// SpatialCNNVQVAE_24781961298459
// MI455X (gfx1250) — matrix operand order repaired and checked by reading, not compiled, not run on hardware
//
#include <hip/hip_runtime.h>
#include <hip/hip_bf16.h>
#include <stdint.h>

typedef __attribute__((ext_vector_type(16))) _Float16 v16h;
typedef __attribute__((ext_vector_type(8)))  float    v8f;

#define WPB 4  // waves per block for GEMM kernels

// K offset for element e of the f16 A-fragment (16x32), lane-half g.
// Contiguous order (matches B): lanes 0-15 -> K=0..15, lanes 16-31 -> K=16..31.
__device__ __forceinline__ int a_kofs(int e, int g) {
    return 16 * g + e;
}

// ---------------------------------------------------------------------------
// 1x1 conv as GEMM:  Y[n,co,p] = sum_ci X[n,ci,p] * W[co,ci] + b[co]
// M = B*HW (pixels), N = Cout, K = Cin.  One wave per 16x16 C tile.
// All geometry compile-time so A/B/store addresses fold to immediate offsets.
// ---------------------------------------------------------------------------
template <int LGHW, int CIN, int COUT, bool RELU>
__global__ void conv1x1_wmma(const float* __restrict__ X, const float* __restrict__ W,
                             const float* __restrict__ Bv, float* __restrict__ Y,
                             long Mtot)
{
    constexpr long HW   = 1L << LGHW;
    constexpr long HWm1 = HW - 1;
    const int wave = threadIdx.x >> 5;
    const int lane = threadIdx.x & 31;
    const long m0 = ((long)blockIdx.x * WPB + wave) * 16;
    if (m0 >= Mtot) return;                    // wave-uniform: EXEC stays all-1s
    const int n0 = blockIdx.y * 16;
    const int g  = lane >> 4;
    const int mr = lane & 15;                  // A row (M) and B column (N)

    const long m    = m0 + mr;
    const long nimg = m >> LGHW;
    const long p    = m & HWm1;
    const float* xr = X + ((nimg * CIN) << LGHW) + p;   // element k at xr[k*HW]
    const float* wc = W + (long)(n0 + mr) * CIN;        // element k at wc[k]

    v8f acc = {};
#pragma unroll
    for (int k0 = 0; k0 < CIN; k0 += 32) {
        v16h af, bf;
#pragma unroll
        for (int e = 0; e < 16; ++e) {
            af[e] = (_Float16)xr[(long)(k0 + a_kofs(e, g)) * HW];
            bf[e] = (_Float16)wc[k0 + 16 * g + e];
        }
        acc = __builtin_amdgcn_wmma_f32_16x16x32_f16(false, af, false, bf,
                                                     (short)0, acc, false, false);
    }
    // C/D: lane -> M = r + 8*g, N = mr.  16-pixel tiles never straddle images.
    const long mo0 = m0 + 8 * g;
    const long ni  = mo0 >> LGHW;
    const long po  = mo0 & HWm1;
    const int  co  = n0 + mr;
    float bb = Bv[co];
    float* yp = Y + ((ni * COUT + co) << LGHW) + po;
#pragma unroll
    for (int r = 0; r < 8; ++r) {
        float v = acc[r] + bb;
        if (RELU) v = fmaxf(v, 0.f);
        yp[r] = v;
    }
}

// ---------------------------------------------------------------------------
// 3x3 conv (pad 1, stride s) as implicit GEMM with K-order k = r9*Cin + ci.
// CIN % 32 == 0, so each 32-wide K chunk has a single (ky,kx): the tap,
// the bounds test and the base pointer are chunk-uniform; only ci varies
// per fragment element (stride IH*IW -> immediate offsets).
// ---------------------------------------------------------------------------
template <int CIN, int LGCIN, int COUT, int IH, int IW, int LGOW, int LGOHW,
          int STRIDE, bool RELU>
__global__ void conv3x3_wmma(const float* __restrict__ X, const float* __restrict__ W,
                             const float* __restrict__ Bv, float* __restrict__ Y,
                             long Mtot)
{
    constexpr int  OW    = 1 << LGOW;
    constexpr int  OWm1  = OW - 1;
    constexpr long OHW   = 1L << LGOHW;
    constexpr long OHWm1 = OHW - 1;
    constexpr long IHW   = (long)IH * IW;
    constexpr int  KTOT  = CIN * 9;

    const int wave = threadIdx.x >> 5;
    const int lane = threadIdx.x & 31;
    const long m0 = ((long)blockIdx.x * WPB + wave) * 16;
    if (m0 >= Mtot) return;
    const int n0 = blockIdx.y * 16;
    const int g  = lane >> 4;
    const int mr = lane & 15;

    const long m    = m0 + mr;
    const long nimg = m >> LGOHW;
    const long rem  = m & OHWm1;
    const int  oy   = (int)(rem >> LGOW);
    const int  ox   = (int)(rem & OWm1);
    const float* wc = W + (long)(n0 + mr) * KTOT;  // OIHW: element (ci,r9) at wc[ci*9+r9]
    const float* xb = X + nimg * CIN * IHW;

    v8f acc = {};
#pragma unroll
    for (int r9 = 0; r9 < 9; ++r9) {
        const int ky = r9 / 3, kx = r9 - 3 * ky;         // compile-time
        const int iy = oy * STRIDE + ky - 1;
        const int ix = ox * STRIDE + kx - 1;
        const bool inb = (iy >= 0) & (iy < IH) & (ix >= 0) & (ix < IW);
        const float* xp = xb + ((long)iy * IW + ix);
#pragma unroll
        for (int ci0 = 0; ci0 < CIN; ci0 += 32) {
            v16h af, bf;
#pragma unroll
            for (int e = 0; e < 16; ++e) {
                int ciA = ci0 + a_kofs(e, g);
                af[e] = (_Float16)(inb ? xp[(long)ciA * IHW] : 0.f);
                int ciB = ci0 + 16 * g + e;
                bf[e] = (_Float16)wc[ciB * 9 + r9];
            }
            acc = __builtin_amdgcn_wmma_f32_16x16x32_f16(false, af, false, bf,
                                                         (short)0, acc, false, false);
        }
    }
    // store: 16-pixel tiles stay within one output row (OW >= 16)
    const long mo0 = m0 + 8 * g;
    const long ni  = mo0 >> LGOHW;
    const long rm  = mo0 & OHWm1;
    const int  co  = n0 + mr;
    float bb = Bv[co];
    float* yp = Y + ((ni * COUT) << LGOHW) + ((long)co << LGOHW) + rm;
#pragma unroll
    for (int r = 0; r < 8; ++r) {
        float v = acc[r] + bb;
        if (RELU) v = fmaxf(v, 0.f);
        yp[r] = v;
    }
}

// ---------------------------------------------------------------------------
// VQ: scores = z . C^T via WMMA; argmin_j ( ||C_j||^2 - 2 z.C_j ).
// One wave handles 16 tokens against all 1024 codes; A fragments (K=128)
// are loaded once, B (codebook) tiles stream from L2.  HW=4096, C=128.
// ---------------------------------------------------------------------------
__global__ void vq_wmma(const float* __restrict__ F, const float* __restrict__ CB,
                        const float* __restrict__ cn, int* __restrict__ idx_i,
                        float* __restrict__ idx_f)
{
    constexpr int LGHW = 12, LGC = 7, NC = 1024;
    constexpr long HW = 1L << LGHW;
    const int lane = threadIdx.x & 31;
    const long m0 = (long)blockIdx.x * 16;
    const int g  = lane >> 4;
    const int mr = lane & 15;

    const long t    = m0 + mr;
    const long nimg = t >> LGHW;
    const long p    = t & (HW - 1);
    const float* fr = F + ((nimg << LGC) << LGHW) + p;

    v16h a[4];
#pragma unroll
    for (int kc = 0; kc < 4; ++kc)
#pragma unroll
        for (int e = 0; e < 16; ++e)
            a[kc][e] = (_Float16)fr[(long)(kc * 32 + a_kofs(e, g)) * HW];

    float bval[8]; int bidx[8];
#pragma unroll
    for (int r = 0; r < 8; ++r) { bval[r] = 3.4e38f; bidx[r] = 0; }

    for (int nt = 0; nt < NC / 16; ++nt) {
        const float* cc = CB + ((long)(nt * 16 + mr) << LGC);
        v8f acc = {};
#pragma unroll
        for (int kc = 0; kc < 4; ++kc) {
            v16h bf;
#pragma unroll
            for (int e = 0; e < 16; ++e)
                bf[e] = (_Float16)cc[kc * 32 + 16 * g + e];
            acc = __builtin_amdgcn_wmma_f32_16x16x32_f16(false, a[kc], false, bf,
                                                         (short)0, acc, false, false);
        }
        int code = nt * 16 + mr;
        float cnv = cn[code];
#pragma unroll
        for (int r = 0; r < 8; ++r) {
            float d = cnv - 2.f * acc[r];
            if (d < bval[r]) { bval[r] = d; bidx[r] = code; }
        }
    }
    // cross-lane min over the 16 code-columns, per token
    __shared__ float sv[2][8][16];
    __shared__ int   si[2][8][16];
#pragma unroll
    for (int r = 0; r < 8; ++r) { sv[g][r][mr] = bval[r]; si[g][r][mr] = bidx[r]; }
    __syncthreads();
    if (lane < 16) {
        int gg = lane >> 3, rr = lane & 7;
        float bv = sv[gg][rr][0]; int bi = si[gg][rr][0];
#pragma unroll
        for (int c2 = 1; c2 < 16; ++c2) {
            float v = sv[gg][rr][c2];
            if (v < bv) { bv = v; bi = si[gg][rr][c2]; }
        }
        long tt = m0 + rr + 8 * gg;
        idx_i[tt] = bi;
        idx_f[tt] = (float)bi;
    }
}

// gather quantized vectors; write quantized_st output ([b,hw,c]) and NCHW
// quant map; accumulate sum((q - z)^2) for vq_loss.  C=128, HW=4096.
__global__ void gather_vq(const float* __restrict__ F, const float* __restrict__ CB,
                          const int* __restrict__ idx, float* __restrict__ outq,
                          float* __restrict__ qmap, float* __restrict__ acc,
                          long Ntok)
{
    constexpr int LGHW = 12, LGC = 7, C = 128;
    long i = (long)blockIdx.x * blockDim.x + threadIdx.x;  // i = t*C + d
    float local = 0.f;
    if (i < (Ntok << LGC)) {
        long t = i >> LGC; int d = (int)(i & (C - 1));
        long nimg = t >> LGHW; long p = t & ((1L << LGHW) - 1);
        long sp = (((nimg << LGC) + d) << LGHW) + p;
        float z = F[sp];
        float q = CB[((long)idx[t] << LGC) + d];
        outq[i] = q;                                   // quantized_st == quantized
        qmap[sp] = q;
        float df = q - z; local = df * df;
    }
    __shared__ float red[256];
    red[threadIdx.x] = local; __syncthreads();
    for (int s = 128; s > 0; s >>= 1) {
        if (threadIdx.x < s) red[threadIdx.x] += red[threadIdx.x + s];
        __syncthreads();
    }
    if (threadIdx.x == 0) atomicAdd(acc, red[0]);
}

// bilinear 4x upsample 64->256, half-pixel centers, edge clamp.
__global__ void bilinear_up(const float* __restrict__ S, float* __restrict__ D,
                            long total)
{
    constexpr int IH = 64, IW = 64, LGOW = 8, LGOHW = 16;
    constexpr int OWm1 = (1 << LGOW) - 1;
    for (long i = (long)blockIdx.x * blockDim.x + threadIdx.x; i < total;
         i += (long)gridDim.x * blockDim.x) {
        long nc  = i >> LGOHW;
        long rem = i & ((1L << LGOHW) - 1);
        int y = (int)(rem >> LGOW), x = (int)(rem & OWm1);
        float sy = (y + 0.5f) * 0.25f - 0.5f;
        float sx = (x + 0.5f) * 0.25f - 0.5f;
        int y0 = (int)floorf(sy); float wy = sy - (float)y0;
        int x0 = (int)floorf(sx); float wx = sx - (float)x0;
        int y1 = y0 + 1, x1 = x0 + 1;
        y0 = max(0, min(IH - 1, y0)); y1 = max(0, min(IH - 1, y1));
        x0 = max(0, min(IW - 1, x0)); x1 = max(0, min(IW - 1, x1));
        const float* s = S + nc * (long)IH * IW;
        float v = (1.f - wy) * ((1.f - wx) * s[y0 * IW + x0] + wx * s[y0 * IW + x1])
                +        wy  * ((1.f - wx) * s[y1 * IW + x0] + wx * s[y1 * IW + x1]);
        D[i] = v;
    }
}

__global__ void codebook_norms(const float* __restrict__ CB, float* __restrict__ cn,
                               int C, int NC)
{
    int j = blockIdx.x * blockDim.x + threadIdx.x;
    if (j < NC) {
        float s = 0.f;
        for (int d = 0; d < C; ++d) { float v = CB[j * C + d]; s += v * v; }
        cn[j] = s;
    }
}

__global__ void sqdiff_reduce(const float* __restrict__ A, const float* __restrict__ Bp,
                              float* __restrict__ acc, long N)
{
    float local = 0.f;
    for (long i = (long)blockIdx.x * blockDim.x + threadIdx.x; i < N;
         i += (long)gridDim.x * blockDim.x) {
        float d = A[i] - Bp[i]; local += d * d;
    }
    __shared__ float red[256];
    red[threadIdx.x] = local; __syncthreads();
    for (int s = 128; s > 0; s >>= 1) {
        if (threadIdx.x < s) red[threadIdx.x] += red[threadIdx.x + s];
        __syncthreads();
    }
    if (threadIdx.x == 0) atomicAdd(acc, red[0]);
}

__global__ void finalize_losses(const float* __restrict__ acc, float* __restrict__ out,
                                float inv_rec, float inv_vq)
{
    float rl = acc[0] * inv_rec;
    float vl = 1.25f * acc[1] * inv_vq;     // (1 + BETA) * mean((q-z)^2)
    out[0] = rl + vl; out[1] = rl; out[2] = vl;
}

// ---------------------------------------------------------------------------
extern "C" void kernel_launch(void* const* d_in, const int* in_sizes, int n_in,
                              void* d_out, int out_size, void* d_ws, size_t ws_size,
                              hipStream_t stream)
{
    (void)in_sizes; (void)n_in; (void)out_size; (void)ws_size;
    const float* x        = (const float*)d_in[0];
    const float* w_spec   = (const float*)d_in[1];
    const float* b_spec   = (const float*)d_in[2];
    const float* w_e1     = (const float*)d_in[3];
    const float* b_e1     = (const float*)d_in[4];
    const float* w_e2     = (const float*)d_in[5];
    const float* b_e2     = (const float*)d_in[6];
    const float* w_proj   = (const float*)d_in[7];
    const float* b_proj   = (const float*)d_in[8];
    const float* codebook = (const float*)d_in[9];
    const float* w_rec    = (const float*)d_in[10];
    const float* b_rec    = (const float*)d_in[11];
    const float* w_fin    = (const float*)d_in[12];
    const float* b_fin    = (const float*)d_in[13];

    float* ws    = (float*)d_ws;
    float* feat1 = ws;                       // [16,128,256,256]; later reused for upsample
    float* feat2 = feat1 + 134217728L;       // [16,128,128,128]
    float* feat3 = feat2 + 33554432L;        // [16,256,64,64]
    float* featP = feat3 + 16777216L;        // [16,128,64,64] tokens
    float* qmap  = featP + 8388608L;         // [16,128,64,64] quantized NCHW
    float* recf  = qmap  + 8388608L;         // [16,128,64,64]
    float* cn    = recf  + 8388608L;         // [1024] codebook norms
    int*   idx   = (int*)(cn + 1024);        // [65536]
    float* acc   = (float*)(idx + 65536);    // [2] {recon_sum, vq_sum}

    float* out_recon = (float*)d_out;                 // [16,64,256,256]
    float* out_q     = out_recon + 67108864L;         // [16,4096,128]
    float* out_idx   = out_q + 8388608L;              // [16,4096] (as float)
    float* out_loss  = out_idx + 65536L;              // [3]

    hipMemsetAsync(acc, 0, 2 * sizeof(float), stream);

    // 1) spec 1x1: 64->128 over 256^2   (M=1048576, N=128, K=64)
    conv1x1_wmma<16, 64, 128, false>
        <<<dim3(16384, 8), 128, 0, stream>>>(x, w_spec, b_spec, feat1, 1048576L);
    // 2) e1 3x3 s2 + relu: 128->128, 256->128
    conv3x3_wmma<128, 7, 128, 256, 256, 7, 14, 2, true>
        <<<dim3(4096, 8), 128, 0, stream>>>(feat1, w_e1, b_e1, feat2, 262144L);
    // 3) e2 3x3 s2 + relu: 128->256, 128->64
    conv3x3_wmma<128, 7, 256, 128, 128, 6, 12, 2, true>
        <<<dim3(1024, 16), 128, 0, stream>>>(feat2, w_e2, b_e2, feat3, 65536L);
    // 4) proj 1x1: 256->128 over 64^2
    conv1x1_wmma<12, 256, 128, false>
        <<<dim3(1024, 8), 128, 0, stream>>>(feat3, w_proj, b_proj, featP, 65536L);
    // 5) codebook squared norms
    codebook_norms<<<4, 256, 0, stream>>>(codebook, cn, 128, 1024);
    // 6) VQ nearest-code search (WMMA scores + argmin)
    vq_wmma<<<4096, 32, 0, stream>>>(featP, codebook, cn, idx, out_idx);
    // 7) gather quantized, write outputs, accumulate vq loss
    gather_vq<<<32768, 256, 0, stream>>>(featP, codebook, idx, out_q, qmap,
                                         acc + 1, 65536L);
    // 8) rec 3x3 s1 + relu: 128->128 over 64^2
    conv3x3_wmma<128, 7, 128, 64, 64, 6, 12, 1, true>
        <<<dim3(1024, 8), 128, 0, stream>>>(qmap, w_rec, b_rec, recf, 65536L);
    // 9) bilinear 4x upsample (reuse feat1 buffer)
    bilinear_up<<<131072, 256, 0, stream>>>(recf, feat1, 134217728L);
    // 10) fin 1x1: 128->64 over 256^2 -> recon output
    conv1x1_wmma<16, 128, 64, false>
        <<<dim3(16384, 4), 128, 0, stream>>>(feat1, w_fin, b_fin, out_recon, 1048576L);
    // 11) recon loss
    sqdiff_reduce<<<8192, 256, 0, stream>>>(out_recon, x, acc, 67108864L);
    // 12) combine losses
    finalize_losses<<<1, 1, 0, stream>>>(acc, out_loss,
                                         1.0f / 67108864.0f, 1.0f / 8388608.0f);
}
